// Net_53137335386395
// MI455X (gfx1250) — compile-verified, matrix operand order checked by reading, not run on hardware
//
#include <hip/hip_runtime.h>

typedef __attribute__((ext_vector_type(16))) _Float16 v16h;
typedef __attribute__((ext_vector_type(8)))  _Float16 v8h;
typedef __attribute__((ext_vector_type(8)))  float    v8f;
typedef __attribute__((ext_vector_type(4)))  int      v4i;

#define BN_EPS 1e-5f

#if __has_builtin(__builtin_amdgcn_global_load_async_to_lds_b128)
#define HAVE_ASYNC_B128 1
// builtin signature (from compiler diagnostic): (AS1 v4i*, AS3 v4i*, imm, imm)
typedef __attribute__((address_space(1))) v4i* gas_v4i_p;
typedef __attribute__((address_space(3))) v4i* las_v4i_p;
#endif

__device__ __forceinline__ void wait_async0() {
#if __has_builtin(__builtin_amdgcn_s_wait_asynccnt)
  __builtin_amdgcn_s_wait_asynccnt(0);
#else
  asm volatile("s_wait_asynccnt 0x0" ::: "memory");
#endif
}

__device__ __forceinline__ v16h cat8(v8h lo, v8h hi) {
  v16h r;
#pragma unroll
  for (int i = 0; i < 8; ++i) { r[i] = lo[i]; r[i + 8] = hi[i]; }
  return r;
}

__device__ __forceinline__ float actq(float v) {
  v = fminf(fmaxf(v, 0.f), 1.f);
  return rintf(v * 255.f) * (1.f / 255.f);   // RNE matches jnp.round
}

// ---------------- stem: Conv2d(3,256,5,p=2)+bias+BN, fp32 direct ----------------
__global__ __launch_bounds__(256) void stem_kernel(
    const float* __restrict__ X, const float* __restrict__ W,
    const float* __restrict__ Bz,
    const float* __restrict__ g, const float* __restrict__ be,
    const float* __restrict__ mu, const float* __restrict__ va,
    float* __restrict__ Y) {
  int idx = blockIdx.x * 256 + threadIdx.x;   // NCHW [256][256][32][32]
  int x = idx & 31; int t = idx >> 5;
  int y = t & 31; t >>= 5;
  int co = t & 255; int b = t >> 8;
  float acc = Bz[co];
  for (int ci = 0; ci < 3; ++ci)
    for (int ky = 0; ky < 5; ++ky) {
      int yy = y + ky - 2;
      if (yy < 0 || yy > 31) continue;
      for (int kx = 0; kx < 5; ++kx) {
        int xx = x + kx - 2;
        if (xx < 0 || xx > 31) continue;
        acc += X[(((size_t)b * 3 + ci) * 32 + yy) * 32 + xx] *
               W[((co * 3 + ci) * 5 + ky) * 5 + kx];
      }
    }
  float inv = g[co] * rsqrtf(va[co] + BN_EPS);
  Y[idx] = acc * inv + (be[co] - mu[co] * inv);
}

// prep: (opt) relu+act_q, channel shuffle, NCHW fp32 -> NHWC fp16
// LDS-transposed so both global read (along pixels) and write (along channels)
// are fully coalesced. Tile = 32 channels x 32 pixels; HW is a multiple of 32,
// so pixel tiles never cross images.
__global__ __launch_bounds__(256) void prep_kernel(
    const float* __restrict__ X, _Float16* __restrict__ Xh,
    int C, int HW, int sg, int doQuant) {
  __shared__ float tile[32][33];
  int t = threadIdx.x;
  int lp = t & 31;          // lane along 32-wide row
  int wr = t >> 5;          // 0..7
  int p0 = blockIdx.x * 32; // flat pixel base (b*HW + hw)
  int c0 = blockIdx.y * 32; // output-channel base
  int b = p0 / HW, hw0 = p0 % HW;
#pragma unroll
  for (int j = 0; j < 4; ++j) {
    int c = c0 + j * 8 + wr;            // output channel (post-shuffle)
    int src = c;
    if (sg > 1) { int gg = c % sg; int m = c / sg; src = gg * (C / sg) + m; }
    tile[j * 8 + wr][lp] = X[((size_t)b * C + src) * HW + hw0 + lp];
  }
  __syncthreads();
#pragma unroll
  for (int j = 0; j < 4; ++j) {
    int p = p0 + j * 8 + wr;
    float v = tile[lp][j * 8 + wr];
    if (doQuant) v = actq(v);
    Xh[(size_t)p * C + c0 + lp] = (_Float16)v;
  }
}

// ------ grouped 1x1 conv as WMMA GEMM: Cin_g=Cout_g=128 for every block -------
// Xh: [B*H*W][Cin] fp16 NHWC, Wq: [Cout][128] fp16, Y: fp32 NCHW.
// Workgroup tile = 64 pixels x 128 out-channels. The shared 64x128 B tile is
// staged into LDS once via async copy (padded row stride 136 halves => the
// ds_load_b128 fragment reads are bank-conflict-free). Each wave computes four
// 16x16 C tiles, reusing its A fragments (held in VGPRs) 4x and amortizing the
// per-channel BN math over the 4 pixel sub-tiles.
// quantMode: 0=none, 1=relu+act_q (quan_mp), 2=relu (last_relu)
__global__ __launch_bounds__(256) void gemm1x1_wmma(
    const _Float16* __restrict__ Xh, const _Float16* __restrict__ Wq,
    const float* __restrict__ bias,
    const float* __restrict__ g, const float* __restrict__ be,
    const float* __restrict__ mu, const float* __restrict__ va,
    float* __restrict__ Y, int Cin, int Cout, int HW, int quantMode) {
  __shared__ _Float16 tileB[64 * 136];    // 64 pixels x 128 K, padded stride
  int tid = threadIdx.x;
  int lane = tid & 31, wave = tid >> 5;
  int nblk = blockIdx.x * 64;             // pixel tile base
  int m0 = blockIdx.y * 128 + wave * 16;  // out-channel tile (8 waves = 128 ch)
  int mi = lane & 15, h = lane >> 4;
  int cinBase = blockIdx.y * 128;         // group*128 (Cin_g==Cout_g==128)

  // stage B tile: 64 rows x 16 chunks of 16B = 1024 chunks, 4 per thread
#pragma unroll
  for (int j = 0; j < 4; ++j) {
    int c = tid + 256 * j;
    int row = c >> 4, col = (c & 15) * 8;
    const _Float16* src = Xh + (size_t)(nblk + row) * Cin + cinBase + col;
    _Float16* dst = tileB + row * 136 + col;
#ifdef HAVE_ASYNC_B128
    __builtin_amdgcn_global_load_async_to_lds_b128(
        (gas_v4i_p)(unsigned long long)(const void*)src,
        (las_v4i_p)(unsigned)(unsigned long long)(void*)dst, 0, 0);
#else
    *(v8h*)dst = *(const v8h*)src;
#endif
  }
#ifdef HAVE_ASYNC_B128
  wait_async0();                          // own copies done before barrier
#endif

  // A fragments: loaded once, reused for all 4 N sub-tiles
  const _Float16* arow = Wq + (size_t)(m0 + mi) * 128;
  __builtin_prefetch(arow, 0, 3);
  v16h afrag[4];
#pragma unroll
  for (int kk = 0; kk < 4; ++kk) {        // K = 128 = 4 x 32
    int kb = kk * 32 + h * 8;             // per-lane 16B chunks at kb, kb+16
    afrag[kk] = cat8(*(const v8h*)(arow + kb), *(const v8h*)(arow + kb + 16));
  }
  __syncthreads();                        // whole tile visible to all 8 waves

  v8f acc[4] = {};
#pragma unroll
  for (int nt = 0; nt < 4; ++nt) {
    const _Float16* browL = tileB + (nt * 16 + mi) * 136;
#pragma unroll
    for (int kk = 0; kk < 4; ++kk) {
      int kb = kk * 32 + h * 8;
      v8h b0 = *(const v8h*)(browL + kb);
      v8h b1 = *(const v8h*)(browL + kb + 16);
      acc[nt] = __builtin_amdgcn_wmma_f32_16x16x32_f16(
          false, afrag[kk], false, cat8(b0, b1), (short)0, acc[nt], false, false);
    }
  }
#pragma unroll
  for (int v = 0; v < 8; ++v) {
    int ch = m0 + v + h * 8;              // C layout: lanes>=16 hold rows v+8
    float inv = g[ch] * rsqrtf(va[ch] + BN_EPS);
    float sh = be[ch] - mu[ch] * inv;
    float bs = bias[ch];
#pragma unroll
    for (int nt = 0; nt < 4; ++nt) {
      int p = nblk + nt * 16 + mi;
      int b = p / HW, hw = p % HW;
      float val = (acc[nt][v] + bs) * inv + sh;
      if (quantMode == 1) val = actq(fmaxf(val, 0.f));
      else if (quantMode == 2) val = fmaxf(val, 0.f);
      Y[((size_t)b * Cout + ch) * HW + hw] = val;
    }
  }
}

// -- grouped 3x3 conv (cin_g=16, cout_g=32) as implicit GEMM, K=144 pad to 160 --
// Wq packed [Cout][160], k = tap*16 + cin (taps row-major ky*3+kx), zeros >=144
__global__ __launch_bounds__(256) void gemm3x3_wmma(
    const _Float16* __restrict__ Xh, const _Float16* __restrict__ Wq,
    const float* __restrict__ bias,
    const float* __restrict__ g, const float* __restrict__ be,
    const float* __restrict__ mu, const float* __restrict__ va,
    float* __restrict__ Y, int Cin, int Cout, int Hh, int Wd) {
  int lane = threadIdx.x & 31, wave = threadIdx.x >> 5;
  int n0 = blockIdx.x * 16;
  int m0 = blockIdx.y * 128 + wave * 16;
  int mi = lane & 15, h = lane >> 4;
  int cinBase = (m0 >> 5) * 16;           // group = m0/32, cin_g = 16
  int HW = Hh * Wd;
  int p = n0 + mi;
  int b = p / HW, hw = p % HW, py = hw / Wd, px = hw % Wd;
  const _Float16* arow = Wq + (size_t)(m0 + mi) * 160;
  v8f acc = {};
#pragma unroll
  for (int kk = 0; kk < 5; ++kk) {        // K = 160 = 5 x 32
    v8h bc0 = {}, bc1 = {};
#pragma unroll
    for (int c2 = 0; c2 < 2; ++c2) {
      int k0 = kk * 32 + h * 8 + c2 * 16; // 16B chunk stays inside one tap
      int tap = k0 >> 4, co = k0 & 15;
      v8h bb = {};
      if (tap < 9) {
        int dy = tap / 3 - 1, dx = tap % 3 - 1;
        int yy = py + dy, xx = px + dx;
        if (yy >= 0 && yy < Hh && xx >= 0 && xx < Wd) {
          size_t pp = ((size_t)b * Hh + yy) * Wd + xx;
          bb = *(const v8h*)(Xh + pp * Cin + cinBase + co);
        }
      }
      if (c2 == 0) bc0 = bb; else bc1 = bb;
    }
    int kb = kk * 32 + h * 8;
    v8h a0 = *(const v8h*)(arow + kb);
    v8h a1 = *(const v8h*)(arow + kb + 16);
    acc = __builtin_amdgcn_wmma_f32_16x16x32_f16(
        false, cat8(a0, a1), false, cat8(bc0, bc1), (short)0, acc, false, false);
  }
#pragma unroll
  for (int v = 0; v < 8; ++v) {
    int ch = m0 + v + h * 8;
    float inv = g[ch] * rsqrtf(va[ch] + BN_EPS);
    float val = (acc[v] + bias[ch]) * inv + (be[ch] - mu[ch] * inv);
    Y[((size_t)b * Cout + ch) * HW + hw] = val;
  }
}

// ---------------------------- 2x2 max pool, fp32 NCHW --------------------------
__global__ __launch_bounds__(256) void maxpool_kernel(
    const float* __restrict__ X, float* __restrict__ Y,
    int C, int Hin, int Win, int total) {
  int idx = blockIdx.x * 256 + threadIdx.x;
  if (idx >= total) return;
  int Ho = Hin >> 1, Wo = Win >> 1;
  int wo = idx % Wo; int t = idx / Wo;
  int ho = t % Ho; t /= Ho;
  int c = t % C; int b = t / C;
  size_t base = (((size_t)b * C + c) * Hin + 2 * ho) * Win + 2 * wo;
  float a = fmaxf(X[base], X[base + 1]);
  float d = fmaxf(X[base + Win], X[base + Win + 1]);
  Y[idx] = fmaxf(a, d);
}

// --------------------- weight prep: max|tanh(w)| reduction --------------------
__global__ __launch_bounds__(256) void maxabs_kernel(
    const float* __restrict__ w, int n, float* __restrict__ outm) {
  __shared__ float s[256];
  float m = 0.f;
  for (int i = blockIdx.x * 256 + threadIdx.x; i < n; i += gridDim.x * 256)
    m = fmaxf(m, fabsf(tanhf(w[i])));
  s[threadIdx.x] = m; __syncthreads();
  for (int off = 128; off > 0; off >>= 1) {
    if (threadIdx.x < off) s[threadIdx.x] = fmaxf(s[threadIdx.x], s[threadIdx.x + off]);
    __syncthreads();
  }
  if (threadIdx.x == 0)
    atomicMax((unsigned int*)outm, __float_as_uint(s[0]));  // positives: bit order == value order
}

__global__ __launch_bounds__(256) void wq1x1_kernel(
    const float* __restrict__ w, _Float16* __restrict__ wq, int n,
    const float* __restrict__ mx) {
  int i = blockIdx.x * 256 + threadIdx.x;
  if (i >= n) return;
  float t = tanhf(w[i]) * (0.5f / mx[0]) + 0.5f;
  float q = rintf(t * 255.f) * (1.f / 255.f);
  wq[i] = (_Float16)(2.f * q - 1.f);
}

// w [cout][16][3][3] -> wq [cout][160], k = tap*16 + cin, zero pad k>=144
__global__ __launch_bounds__(256) void wq3x3_kernel(
    const float* __restrict__ w, _Float16* __restrict__ wq, int total,
    const float* __restrict__ mx) {
  int i = blockIdx.x * 256 + threadIdx.x;
  if (i >= total) return;
  int k = i % 160; int o = i / 160;
  float valf = 0.f;
  if (k < 144) {
    int ci = k & 15, tap = k >> 4;
    float t = tanhf(w[(o * 16 + ci) * 9 + tap]) * (0.5f / mx[0]) + 0.5f;
    float q = rintf(t * 255.f) * (1.f / 255.f);
    valf = 2.f * q - 1.f;
  }
  wq[i] = (_Float16)valf;
}

// ---------------- head: 1x1 conv 1024->10 + BN + relu + mean(8x8) --------------
__global__ __launch_bounds__(640) void head_kernel(
    const float* __restrict__ X, const float* __restrict__ W,
    const float* __restrict__ Bz,
    const float* __restrict__ g, const float* __restrict__ be,
    const float* __restrict__ mu, const float* __restrict__ va,
    float* __restrict__ out) {
  __shared__ float s[640];
  int b = blockIdx.x;
  int tid = threadIdx.x;
  int cls = tid >> 6, px = tid & 63;
  const float* xb = X + (size_t)b * 1024 * 64 + px;
  const float* wr = W + cls * 1024;
  float acc = Bz[cls];
  for (int c = 0; c < 1024; ++c) acc += xb[(size_t)c * 64] * wr[c];
  float inv = g[cls] * rsqrtf(va[cls] + BN_EPS);
  acc = fmaxf(acc * inv + (be[cls] - mu[cls] * inv), 0.f);
  s[tid] = acc; __syncthreads();
  for (int off = 32; off > 0; off >>= 1) {
    if (px < off) s[tid] += s[tid + off];
    __syncthreads();
  }
  if (px == 0) out[b * 10 + cls] = s[tid] * (1.f / 64.f);
}

extern "C" void kernel_launch(void* const* d_in, const int* in_sizes, int n_in,
                              void* d_out, int out_size, void* d_ws, size_t ws_size,
                              hipStream_t stream) {
  (void)in_sizes; (void)n_in; (void)out_size; (void)ws_size;
  const float* x    = (const float*)d_in[0];
  const float* w0   = (const float*)d_in[1];
  const float* b0   = (const float*)d_in[2];
  const float* bn0g = (const float*)d_in[3];
  const float* bn0b = (const float*)d_in[4];
  const float* bn0m = (const float*)d_in[5];
  const float* bn0v = (const float*)d_in[6];
  const float *cw[7], *cb[7], *bg[7], *bb[7], *bm[7], *bv[7];
  for (int i = 0; i < 7; ++i) {
    cw[i] = (const float*)d_in[7 + 6 * i];
    cb[i] = (const float*)d_in[8 + 6 * i];
    bg[i] = (const float*)d_in[9 + 6 * i];
    bb[i] = (const float*)d_in[10 + 6 * i];
    bm[i] = (const float*)d_in[11 + 6 * i];
    bv[i] = (const float*)d_in[12 + 6 * i];
  }
  const float* wf   = (const float*)d_in[49];
  const float* bf   = (const float*)d_in[50];
  const float* bnfg = (const float*)d_in[51];
  const float* bnfb = (const float*)d_in[52];
  const float* bnfm = (const float*)d_in[53];
  const float* bnfv = (const float*)d_in[54];

  // workspace layout
  char* ws = (char*)d_ws;
  float*    bufA = (float*)ws;                                   // 256 MiB fp32 NCHW
  float*    bufB = (float*)(ws + (size_t)268435456);             // 256 MiB fp32 NCHW
  _Float16* bufH = (_Float16*)(ws + (size_t)536870912);          // 128 MiB fp16 NHWC
  _Float16* wqb  = (_Float16*)(ws + (size_t)536870912 + 134217728);
  const size_t wqoff[7] = {0, 32768, 65536, 147456, 212992, 278528, 442368};
  _Float16* wq[7];
  for (int i = 0; i < 7; ++i) wq[i] = wqb + wqoff[i];
  float* maxb = (float*)((char*)wqb + 2 * 573440);

  (void)hipMemsetAsync(maxb, 0, 7 * sizeof(float), stream);

  // weight quantization (tanh scale reduction + 8-bit quantize to fp16, packed)
  const int wn[7] = {32768, 32768, 73728, 65536, 65536, 147456, 131072};
  for (int i = 0; i < 7; ++i)
    maxabs_kernel<<<256, 256, 0, stream>>>(cw[i], wn[i], maxb + i);
  wq1x1_kernel<<<128, 256, 0, stream>>>(cw[0], wq[0], 32768, maxb + 0);
  wq1x1_kernel<<<128, 256, 0, stream>>>(cw[1], wq[1], 32768, maxb + 1);
  wq3x3_kernel<<<320, 256, 0, stream>>>(cw[2], wq[2], 81920, maxb + 2);
  wq1x1_kernel<<<256, 256, 0, stream>>>(cw[3], wq[3], 65536, maxb + 3);
  wq1x1_kernel<<<256, 256, 0, stream>>>(cw[4], wq[4], 65536, maxb + 4);
  wq3x3_kernel<<<640, 256, 0, stream>>>(cw[5], wq[5], 163840, maxb + 5);
  wq1x1_kernel<<<512, 256, 0, stream>>>(cw[6], wq[6], 131072, maxb + 6);

  // stem
  stem_kernel<<<262144, 256, 0, stream>>>(x, w0, b0, bn0g, bn0b, bn0m, bn0v, bufA);

  // block 0: quan_nor, no shuffle, G=2, 256->256 @32x32
  prep_kernel<<<dim3(8192, 8), 256, 0, stream>>>(bufA, bufH, 256, 1024, 1, 1);
  gemm1x1_wmma<<<dim3(4096, 2), 256, 0, stream>>>(bufH, wq[0], cb[0],
      bg[0], bb[0], bm[0], bv[0], bufB, 256, 256, 1024, 0);
  // block 1: quan_nor, shuffle g=2, quan_mp, then pool 32->16
  prep_kernel<<<dim3(8192, 8), 256, 0, stream>>>(bufB, bufH, 256, 1024, 2, 1);
  gemm1x1_wmma<<<dim3(4096, 2), 256, 0, stream>>>(bufH, wq[1], cb[1],
      bg[1], bb[1], bm[1], bv[1], bufA, 256, 256, 1024, 1);
  maxpool_kernel<<<65536, 256, 0, stream>>>(bufA, bufB, 256, 32, 32, 16777216);
  // block 2: 3x3 G=16, shuffle g=2, 256->512 @16x16
  prep_kernel<<<dim3(2048, 8), 256, 0, stream>>>(bufB, bufH, 256, 256, 2, 0);
  gemm3x3_wmma<<<dim3(4096, 4), 256, 0, stream>>>(bufH, wq[2], cb[2],
      bg[2], bb[2], bm[2], bv[2], bufA, 256, 512, 16, 16);
  // block 3: quan_nor, shuffle g=16, G=4, 512->512 @16x16
  prep_kernel<<<dim3(2048, 16), 256, 0, stream>>>(bufA, bufH, 512, 256, 16, 1);
  gemm1x1_wmma<<<dim3(1024, 4), 256, 0, stream>>>(bufH, wq[3], cb[3],
      bg[3], bb[3], bm[3], bv[3], bufB, 512, 512, 256, 0);
  // block 4: quan_nor, shuffle g=4, quan_mp, then pool 16->8
  prep_kernel<<<dim3(2048, 16), 256, 0, stream>>>(bufB, bufH, 512, 256, 4, 1);
  gemm1x1_wmma<<<dim3(1024, 4), 256, 0, stream>>>(bufH, wq[4], cb[4],
      bg[4], bb[4], bm[4], bv[4], bufA, 512, 512, 256, 1);
  maxpool_kernel<<<32768, 256, 0, stream>>>(bufA, bufB, 512, 16, 16, 8388608);
  // block 5: 3x3 G=32, shuffle g=4, 512->1024 @8x8
  prep_kernel<<<dim3(512, 16), 256, 0, stream>>>(bufB, bufH, 512, 64, 4, 0);
  gemm3x3_wmma<<<dim3(1024, 8), 256, 0, stream>>>(bufH, wq[5], cb[5],
      bg[5], bb[5], bm[5], bv[5], bufA, 512, 1024, 8, 8);
  // block 6: quan_nor, shuffle g=32, G=8, 1024->1024 @8x8, last_relu
  prep_kernel<<<dim3(512, 32), 256, 0, stream>>>(bufA, bufH, 1024, 64, 32, 1);
  gemm1x1_wmma<<<dim3(256, 8), 256, 0, stream>>>(bufH, wq[6], cb[6],
      bg[6], bb[6], bm[6], bv[6], bufB, 1024, 1024, 64, 2);
  // head
  head_kernel<<<256, 640, 0, stream>>>(bufB, wf, bf, bnfg, bnfb, bnfm, bnfv,
                                       (float*)d_out);
}
